// EncoderCell_78486232367525
// MI455X (gfx1250) — compile-verified, matrix operand order checked by reading, not run on hardware
//
#include <hip/hip_runtime.h>
#include <hip/hip_bf16.h>

#define BB 16
#define TT 49
#define NN 64
#define HH 128
#define MM (BB*NN)      // 1024 rows per (g,t) GEMM
#define K1 (5*HH)       // 640 fused K for gate GEMM
#define NG 8

typedef __bf16 bf16_t;
typedef __attribute__((ext_vector_type(16))) __bf16 v16bf;
typedef __attribute__((ext_vector_type(8)))  float  v8f;
typedef __attribute__((ext_vector_type(4)))  int    v4i;

__device__ __forceinline__ float fsig(float x){ return 1.0f/(1.0f+__expf(-x)); }

// ---------------------------------------------------------------------------
// Pack activations into bf16 A[T][1024][640] = [p | h(t-1) | h | h(t+1) | h(n-1)]
// ---------------------------------------------------------------------------
__global__ void pack_A(const float* __restrict__ p, const float* __restrict__ h,
                       bf16_t* __restrict__ A){
  long idx = (long)blockIdx.x*blockDim.x + threadIdx.x;
  const long total = (long)TT*MM*K1;
  if (idx >= total) return;
  int j = (int)(idx % K1);
  long r = idx / K1;
  int m = (int)(r % MM);
  int t = (int)(r / MM);
  int b = m >> 6, n = m & 63;
  int seg = j >> 7, k = j & 127;
  float v = 0.f;
  if (seg == 0)                  v = p[(((long)(b*TT+t)*NN + n)<<7) + k];
  else if (seg == 1){ if (t>0)   v = h[(((long)(b*TT+t-1)*NN + n)<<7) + k]; }
  else if (seg == 2)             v = h[(((long)(b*TT+t)*NN + n)<<7) + k];
  else if (seg == 3){ if (t<TT-1)v = h[(((long)(b*TT+t+1)*NN + n)<<7) + k]; }
  else              { if (n>0)   v = h[(((long)(b*TT+t)*NN + n-1)<<7) + k]; }
  A[idx] = (bf16_t)v;
}

// ---------------------------------------------------------------------------
// Transpose-pack weights to bf16 BT[g][t][kout=128][j=640] (output-major),
// j concatenates U(128) | Wt(384) | Ws(128).  32x32 LDS tile transpose.
// ---------------------------------------------------------------------------
__global__ void pack_BT(const float* __restrict__ U, const float* __restrict__ Wt,
                        const float* __restrict__ Ws, bf16_t* __restrict__ BT){
  __shared__ float tile[32][33];
  int z  = blockIdx.z;            // g*TT + t
  int j0 = blockIdx.x*32;         // input-dim tile (0..640)
  int k0 = blockIdx.y*32;         // output-dim tile (0..128)
  int tx = threadIdx.x & 31, ty = threadIdx.x >> 5;   // 256 thr: ty 0..7
  #pragma unroll
  for (int i=0;i<4;++i){
    int j = j0 + ty + i*8;
    int k = k0 + tx;
    float v;
    if (j < HH)        v = U [((long)z*HH   + j)       *HH + k];
    else if (j < 4*HH) v = Wt[((long)z*3*HH + (j-HH))  *HH + k];
    else               v = Ws[((long)z*HH   + (j-4*HH))*HH + k];
    tile[ty+i*8][tx] = v;
  }
  __syncthreads();
  #pragma unroll
  for (int i=0;i<4;++i){
    int ko = k0 + ty + i*8;
    int j  = j0 + tx;
    BT[((long)z*HH + ko)*K1 + j] = (bf16_t)tile[tx][ty+i*8];
  }
}

// Transpose-pack Wg[0] to bf16 BgT[t][kout][h]
__global__ void pack_BgT(const float* __restrict__ Wg0, bf16_t* __restrict__ BgT){
  __shared__ float tile[32][33];
  int t  = blockIdx.z;
  int j0 = blockIdx.x*32, k0 = blockIdx.y*32;
  int tx = threadIdx.x & 31, ty = threadIdx.x >> 5;
  #pragma unroll
  for (int i=0;i<4;++i)
    tile[ty+i*8][tx] = Wg0[((long)t*HH + (j0+ty+i*8))*HH + (k0+tx)];
  __syncthreads();
  #pragma unroll
  for (int i=0;i<4;++i)
    BgT[((long)t*HH + (k0+ty+i*8))*HH + (j0+tx)] = (bf16_t)tile[tx][ty+i*8];
}

// ---------------------------------------------------------------------------
// GEMV: pre[g][b][t][k] = g_t[b,t,:] @ Z[g,t] + bias[g,t,k]   (broadcast terms)
// blockIdx.x = (g*BB + b)*TT + t ; 128 threads = k
// ---------------------------------------------------------------------------
__global__ void gemv_zt(const float* __restrict__ gt, const float* __restrict__ Z,
                        const float* __restrict__ bias, float* __restrict__ pre){
  int k  = threadIdx.x;
  int t  = blockIdx.x % TT;
  int gb = blockIdx.x / TT;
  int b  = gb % BB;
  int g  = gb / BB;
  __shared__ float gs[HH];
  gs[k] = gt[((long)b*TT + t)*HH + k];
  __syncthreads();
  const float* W = Z + ((long)g*TT + t)*HH*HH;
  float acc = bias[((long)g*TT + t)*HH + k];
  for (int hh=0; hh<HH; ++hh) acc += gs[hh]*W[hh*HH + k];
  pre[(long)blockIdx.x*HH + k] = acc;
}

// ---------------------------------------------------------------------------
// Main bf16 WMMA GEMM:  out[g][t][m][k] = act( A[t] @ BT[g][t]^T + bcast )
// grid (M/128, T, G), 256 threads = 8 wave32s; wave -> 16x128 output strip.
// B tile staged via double-buffered GLOBAL_LOAD_ASYNC_TO_LDS_B128 (ASYNCcnt).
// mode 0: gates (sigmoid for g<7, tanh for g==7) ; mode 1: sigmoid all.
// ---------------------------------------------------------------------------
__global__ void gemm_bf16(const bf16_t* __restrict__ A, const bf16_t* __restrict__ BT,
                          const float* __restrict__ bcast, float* __restrict__ out,
                          int K, int mode){
  __shared__ __align__(32) bf16_t lb[2][HH][48]; // 32 cols used + pad (32B rows)

  int g = blockIdx.z, t = blockIdx.y, mtile = blockIdx.x;
  const bf16_t* Ab = A  + (long)t*MM*K;
  const bf16_t* Bb = BT + ((long)g*TT + t)*HH*K;
  float*        Ob = out + ((long)g*TT + t)*MM*HH;
  const float*  Cb = bcast + (long)g*BB*TT*HH;

  int tid  = threadIdx.x;
  int lane = tid & 31, wave = tid >> 5;
  int mbase = mtile*128 + wave*16;
  int arow  = mbase + (lane & 15);
  int ksel  = (lane >> 4) * 16;          // lanes 0-15 -> K 0..15, 16-31 -> K 16..31

  v8f acc[8];
  #pragma unroll
  for (int c=0;c<8;++c){ v8f z; for (int e=0;e<8;++e) z[e]=0.f; acc[c]=z; }

  // Each thread stages 32B (two async b128) of the 128x32 bf16 B tile.
  int r = tid >> 1, hf = tid & 1;
  const bf16_t* gB = Bb + (long)r*K + hf*16;
  unsigned ldsA = (unsigned)(size_t)&lb[0][r][hf*16];
  unsigned ldsB = (unsigned)(size_t)&lb[1][r][hf*16];

  auto issue_tile = [&](int buf, int kb){
    const bf16_t* gp = gB + kb;
    unsigned la = buf ? ldsB : ldsA;
    asm volatile("global_load_async_to_lds_b128 %0, %1, off"
                 :: "v"(la), "v"(gp) : "memory");
    asm volatile("global_load_async_to_lds_b128 %0, %1, off offset:16"
                 :: "v"(la), "v"(gp) : "memory");
  };

  const int nIter = K >> 5;
  issue_tile(0, 0);
  for (int i = 0; i < nIter; ++i){
    int kb = i << 5;
    if (i + 1 < nIter){
      issue_tile((i + 1) & 1, kb + 32);                 // prefetch next tile
      asm volatile("s_wait_asynccnt 0x2" ::: "memory"); // current tile done
      __builtin_prefetch(Ab + (long)arow*K + kb + 32 + ksel, 0, 1);
    } else {
      asm volatile("s_wait_asynccnt 0x0" ::: "memory");
    }
    __syncthreads();                                    // tile visible to all waves

    const bf16_t (*cur)[48] = lb[i & 1];
    v16bf af = *(const v16bf*)(Ab + (long)arow*K + kb + ksel);
    #pragma unroll
    for (int c=0;c<8;++c){
      v16bf bf = *(const v16bf*)(&cur[c*16 + (lane & 15)][ksel]);
      acc[c] = __builtin_amdgcn_wmma_f32_16x16x32_bf16(
                 false, af, false, bf, (short)0, acc[c], false, false);
    }
    __syncthreads();     // all waves done reading before this buffer is refilled
  }

  // Epilogue: C/D layout -> VGPR j: lanes 0-15 M=j, lanes 16-31 M=j+8
  int col0 = lane & 15;
  int rhi  = (lane >> 4) * 8;
  bool dotanh = (mode == 0 && g == 7);
  #pragma unroll
  for (int c=0;c<8;++c){
    int col = c*16 + col0;
    #pragma unroll
    for (int j=0;j<8;++j){
      int row = mbase + j + rhi;
      int b = row >> 6;
      float v = acc[c][j] + Cb[((long)b*TT + t)*HH + col];
      v = dotanh ? tanhf(v) : fsig(v);
      Ob[(long)row*HH + col] = v;
    }
  }
}

// ---------------------------------------------------------------------------
// Combine 8 activated gates into c_h_new / h_new; emit bf16 h_new for GEMM2.
// ---------------------------------------------------------------------------
__global__ void combine1(const float* __restrict__ gates, const float* __restrict__ c_h,
                         const float* __restrict__ c_gt,
                         float* __restrict__ out_h, float* __restrict__ out_ch,
                         bf16_t* __restrict__ A2){
  long idx = (long)blockIdx.x*blockDim.x + threadIdx.x;
  const long total = (long)BB*TT*NN*HH;
  if (idx >= total) return;
  int k = (int)(idx & 127);
  long r = idx >> 7;
  int n = (int)(r & 63); r >>= 6;
  int t = (int)(r % TT); int b = (int)(r / TT);

  const long gstride = (long)TT*MM*HH;
  long gidx = ((long)t*MM + (b*NN + n))*HH + k;
  float gi   = gates[0*gstride + gidx];
  float f_lt = gates[1*gstride + gidx];
  float f_ft = gates[2*gstride + gidx];
  float f_rt = gates[3*gstride + gidx];
  float f_s  = gates[4*gstride + gidx];
  float gg   = gates[5*gstride + gidx];
  float oo   = gates[6*gstride + gidx];
  float cc   = gates[7*gstride + gidx];

  float ch  = c_h[idx];
  float ctb = (t > 0)    ? c_h[idx - (long)NN*HH] : 0.f;
  float cta = (t < TT-1) ? c_h[idx + (long)NN*HH] : 0.f;
  float csb = (n > 0)    ? c_h[idx - HH]          : 0.f;
  float cgt = c_gt[((long)b*TT + t)*HH + k];

  float chn = gi*cc + f_lt*ctb + f_ft*ch + f_rt*cta + f_s*csb + gg*cgt;
  float hn  = oo * tanhf(chn);
  out_h[idx]  = hn;
  out_ch[idx] = chn;
  A2[gidx] = (bf16_t)hn;
}

// mean over bones: hm[b][t][k]
__global__ void hmean_k(const float* __restrict__ hnew, float* __restrict__ hm){
  int k = threadIdx.x;
  const float* p0 = hnew + (long)blockIdx.x*NN*HH + k;
  float s = 0.f;
  for (int n=0;n<NN;++n) s += p0[n*HH];
  hm[(long)blockIdx.x*HH + k] = s * (1.0f/NN);
}

// g_gtg / o_gto = sigmoid(h_mean @ Wg[1|2] + zg_pre[1|2]); blockIdx.x = b*TT+t
__global__ void gates_g(const float* __restrict__ hm, const float* __restrict__ Wg,
                        const float* __restrict__ zgpre, float* __restrict__ ggv,
                        float* __restrict__ oov){
  int k = threadIdx.x;
  int t = blockIdx.x % TT;
  __shared__ float hs[HH];
  hs[k] = hm[(long)blockIdx.x*HH + k];
  __syncthreads();
  const float* W1 = Wg + ((long)(1*TT + t))*HH*HH;
  const float* W2 = Wg + ((long)(2*TT + t))*HH*HH;
  float a1 = zgpre[((long)(1*BB*TT) + blockIdx.x)*HH + k];
  float a2 = zgpre[((long)(2*BB*TT) + blockIdx.x)*HH + k];
  for (int hh=0; hh<HH; ++hh){
    float hv = hs[hh];
    a1 += hv * W1[hh*HH + k];
    a2 += hv * W2[hh*HH + k];
  }
  ggv[(long)blockIdx.x*HH + k] = fsig(a1);
  oov[(long)blockIdx.x*HH + k] = fsig(a2);
}

// c_g_new = mean_n(f_gtf * c_h_new) + g_gtg*c_g_t ; g_new = tanh(c_g_new)*o_gto
__global__ void final_g(const float* __restrict__ fact, const float* __restrict__ ch_new,
                        const float* __restrict__ cgt, const float* __restrict__ ggv,
                        const float* __restrict__ oov, float* __restrict__ out_g,
                        float* __restrict__ out_cg){
  int k = threadIdx.x;
  int t = blockIdx.x % TT;
  int b = blockIdx.x / TT;
  const float* fb = fact   + ((long)t*MM + b*NN)*HH + k;   // fact layout [t][m][k]
  const float* cb = ch_new + (long)blockIdx.x*NN*HH + k;
  float s = 0.f;
  for (int n=0;n<NN;++n) s += fb[n*HH] * cb[n*HH];
  s *= (1.0f/NN);
  float cg = s + ggv[(long)blockIdx.x*HH + k]*cgt[(long)blockIdx.x*HH + k];
  out_cg[(long)blockIdx.x*HH + k] = cg;
  out_g [(long)blockIdx.x*HH + k] = tanhf(cg)*oov[(long)blockIdx.x*HH + k];
}

// ---------------------------------------------------------------------------
extern "C" void kernel_launch(void* const* d_in, const int* in_sizes, int n_in,
                              void* d_out, int out_size, void* d_ws, size_t ws_size,
                              hipStream_t stream) {
  const float* h   = (const float*)d_in[0];
  const float* c_h = (const float*)d_in[1];
  const float* p   = (const float*)d_in[2];
  const float* g_t = (const float*)d_in[3];
  const float* c_g = (const float*)d_in[4];
  const float* U   = (const float*)d_in[5];
  const float* Wt  = (const float*)d_in[6];
  const float* Ws  = (const float*)d_in[7];
  const float* Zt  = (const float*)d_in[8];
  const float* bb  = (const float*)d_in[9];
  const float* Wg  = (const float*)d_in[10];
  const float* Zg  = (const float*)d_in[11];
  const float* bg  = (const float*)d_in[12];

  float* out_h  = (float*)d_out;
  float* out_ch = out_h  + (long)BB*TT*NN*HH;
  float* out_g  = out_ch + (long)BB*TT*NN*HH;
  float* out_cg = out_g  + (long)BB*TT*HH;

  char* wsb = (char*)d_ws;
  auto carve = [&](size_t bytes)->char*{
    char* r = wsb; wsb += (bytes + 255) & ~(size_t)255; return r;
  };
  bf16_t* A1   = (bf16_t*)carve((size_t)TT*MM*K1*2);        // packed activations
  bf16_t* BT1  = (bf16_t*)carve((size_t)NG*TT*HH*K1*2);     // packed weights^T
  bf16_t* BgT  = (bf16_t*)carve((size_t)TT*HH*HH*2);        // Wg[0]^T
  float*  pre2 = (float*) carve((size_t)NG*BB*TT*HH*4);     // Zt term + bias
  float*  zgp  = (float*) carve((size_t)3*BB*TT*HH*4);      // Zg terms + bg
  float*  gts  = (float*) carve((size_t)NG*TT*MM*HH*4);     // activated gates
  bf16_t* A2   = (bf16_t*)carve((size_t)TT*MM*HH*2);        // bf16 h_new
  float*  hm   = (float*) carve((size_t)BB*TT*HH*4);
  float*  ggv  = (float*) carve((size_t)BB*TT*HH*4);
  float*  oov  = (float*) carve((size_t)BB*TT*HH*4);
  float*  fac  = (float*) carve((size_t)TT*MM*HH*4);        // sigmoid(f_gtf)

  long nA = (long)TT*MM*K1;
  pack_A  <<<(unsigned)((nA+255)/256), 256, 0, stream>>>(p, h, A1);
  pack_BT <<<dim3(K1/32, HH/32, NG*TT), 256, 0, stream>>>(U, Wt, Ws, BT1);
  pack_BgT<<<dim3(HH/32, HH/32, TT),   256, 0, stream>>>(Wg, BgT);
  gemv_zt <<<NG*BB*TT, HH, 0, stream>>>(g_t, Zt, bb, pre2);
  gemv_zt <<<3*BB*TT,  HH, 0, stream>>>(g_t, Zg, bg, zgp);

  gemm_bf16<<<dim3(MM/128, TT, NG), 256, 0, stream>>>(A1, BT1, pre2, gts, K1, 0);

  long nC = (long)BB*TT*NN*HH;
  combine1<<<(unsigned)((nC+255)/256), 256, 0, stream>>>(gts, c_h, c_g, out_h, out_ch, A2);
  hmean_k <<<BB*TT, HH, 0, stream>>>(out_h, hm);
  gates_g <<<BB*TT, HH, 0, stream>>>(hm, Wg, zgp, ggv, oov);

  gemm_bf16<<<dim3(MM/128, TT, 1), 256, 0, stream>>>(A2, BgT, zgp, fac, HH, 1);
  final_g <<<BB*TT, HH, 0, stream>>>(fac, out_ch, c_g, ggv, oov, out_g, out_cg);
}
